// CrossAttentionBlock_35278861370042
// MI455X (gfx1250) — compile-verified, matrix operand order checked by reading, not run on hardware
//
#include <hip/hip_runtime.h>

// ---------------------------------------------------------------- types
typedef __bf16 bf16_t;
typedef __attribute__((ext_vector_type(16))) bf16_t v16bf;
typedef __attribute__((ext_vector_type(8)))  bf16_t v8bf;
typedef __attribute__((ext_vector_type(8)))  float  v8f;

// ---------------------------------------------------------------- shapes
constexpr int Bc   = 2;
constexpr int Sc   = 2048;
constexpr int Dc   = 2048;
constexpr int Hc   = 16;
constexpr int HDc  = 128;     // head dim
constexpr int HIDc = 5632;
constexpr float EPSc = 1e-5f;

// ================================================================ device helpers
__device__ __forceinline__ v16bf load_frag(const bf16_t* rowbase, int half) {
  // ISA 16-bit A/B fragment: elements 0..7  -> K = 8*half + e
  //                          elements 8..15 -> K = 16 + 8*half + (e-8)
  v8bf lo = *(const v8bf*)(rowbase + 8 * half);
  v8bf hi = *(const v8bf*)(rowbase + 16 + 8 * half);
  v16bf f;
#pragma unroll
  for (int i = 0; i < 8; ++i) { f[i] = lo[i]; f[i + 8] = hi[i]; }
  return f;
}

__device__ __forceinline__ v8f wmma_bf16(v16bf a, v16bf b, v8f c) {
  return __builtin_amdgcn_wmma_f32_16x16x32_bf16(
      /*neg_a=*/false, a, /*neg_b=*/false, b,
      /*c_mod=*/(short)0, c, /*reuse_a=*/false, /*reuse_b=*/false);
}

// Async global->LDS DMA (CDNA5), GV mode: per-lane 64-bit global address,
// per-lane 32-bit LDS offset. Tracked by ASYNCcnt.
__device__ __forceinline__ unsigned lds_off(const void* p) {
  return (unsigned)(unsigned long long)p;   // low 32 bits of generic LDS ptr
}
__device__ __forceinline__ void async_g2l_b128(unsigned lds_addr, const void* gptr) {
  asm volatile("global_load_async_to_lds_b128 %0, %1, off"
               :: "v"(lds_addr), "v"((unsigned long long)gptr)
               : "memory");
}
__device__ __forceinline__ void wait_async0() {
  asm volatile("s_wait_asynccnt 0x0" ::: "memory");
}

// ================================================================ RMSNorm (row = 2048)
template <bool BF16OUT, bool RESID>
__global__ __launch_bounds__(256) void rmsnorm_kernel(
    const float* __restrict__ x, const float* __restrict__ resid,
    const float* __restrict__ w, void* __restrict__ out) {
  const int row = blockIdx.x;
  const float* xr = x + (size_t)row * Dc;
  const float* rr = RESID ? (resid + (size_t)row * Dc) : nullptr;
  __shared__ float red[256];
  float vals[8];
  float ss = 0.f;
#pragma unroll
  for (int i = 0; i < 8; ++i) {
    int c = threadIdx.x + i * 256;
    float v = xr[c];
    if (RESID) v += rr[c];
    vals[i] = v;
    ss += v * v;
  }
  red[threadIdx.x] = ss;
  __syncthreads();
  for (int s = 128; s > 0; s >>= 1) {
    if ((int)threadIdx.x < s) red[threadIdx.x] += red[threadIdx.x + s];
    __syncthreads();
  }
  float inv = rsqrtf(red[0] / (float)Dc + EPSc);
#pragma unroll
  for (int i = 0; i < 8; ++i) {
    int c = threadIdx.x + i * 256;
    float o = vals[i] * inv * w[c];
    if (BF16OUT) ((bf16_t*)out)[(size_t)row * Dc + c] = (bf16_t)o;
    else         ((float*)out)[(size_t)row * Dc + c]  = o;
  }
}

// ================================================================ weight transpose+cast: f32 KxN -> bf16 NxK
__global__ __launch_bounds__(256) void tcast_kernel(
    const float* __restrict__ W, void* __restrict__ Wt_, int K, int N) {
  bf16_t* Wt = (bf16_t*)Wt_;
  __shared__ float tile[32][33];
  const int kb = blockIdx.x * 32, nb = blockIdx.y * 32;
  const int tx = threadIdx.x & 31, ty = threadIdx.x >> 5;
#pragma unroll
  for (int i = 0; i < 32; i += 8)
    tile[ty + i][tx] = W[(size_t)(kb + ty + i) * N + nb + tx];
  __syncthreads();
#pragma unroll
  for (int i = 0; i < 32; i += 8)
    Wt[(size_t)(nb + ty + i) * K + kb + tx] = (bf16_t)tile[tx][ty + i];
}

// ================================================================ bf16 transpose (B,S,H,HD) -> (B,H,HD,S)
__global__ __launch_bounds__(256) void vtile_kernel(
    const void* __restrict__ vin_, void* __restrict__ vt_) {
  const bf16_t* vin = (const bf16_t*)vin_;
  bf16_t* vt = (bf16_t*)vt_;
  __shared__ bf16_t tile[32][33];
  const int bh = blockIdx.z;           // b*H + h
  const int b = bh >> 4, h = bh & 15;
  const int s0 = blockIdx.x * 32, d0 = blockIdx.y * 32;
  const int tx = threadIdx.x & 31, ty = threadIdx.x >> 5;
#pragma unroll
  for (int i = 0; i < 32; i += 8)
    tile[ty + i][tx] = vin[((size_t)(b * Sc + s0 + ty + i)) * Dc + h * HDc + d0 + tx];
  __syncthreads();
#pragma unroll
  for (int i = 0; i < 32; i += 8)
    vt[((size_t)bh * HDc + d0 + ty + i) * Sc + s0 + tx] = tile[tx][ty + i];
}

// ================================================================ RoPE + reshape (B,S,H,HD) -> (B,H,S,HD), optional scale
__global__ __launch_bounds__(256) void rope_kernel(
    const void* __restrict__ lin_, const float* __restrict__ cs,
    const float* __restrict__ sn, void* __restrict__ out_, float scale) {
  const bf16_t* lin = (const bf16_t*)lin_;
  bf16_t* out = (bf16_t*)out_;
  const int idx = blockIdx.x * 256 + threadIdx.x;  // (b,s,h,j) j in [0,64)
  const int j = idx & 63;
  const int h = (idx >> 6) & 15;
  const int s = (idx >> 10) & 2047;
  const int b = idx >> 21;
  const size_t bi = ((size_t)(b * Sc + s)) * Dc + h * HDc;
  float x1 = (float)lin[bi + 2 * j];
  float x2 = (float)lin[bi + 2 * j + 1];
  float c = cs[(size_t)s * 64 + j];
  float si = sn[(size_t)s * 64 + j];
  float o1 = (x1 * c - x2 * si) * scale;
  float o2 = (x1 * si + x2 * c) * scale;
  const size_t bo = (((size_t)(b * Hc + h)) * Sc + s) * HDc;
  out[bo + 2 * j] = (bf16_t)o1;
  out[bo + 2 * j + 1] = (bf16_t)o2;
}

// ================================================================ WMMA GEMM  C[M,N] = A[M,K](bf16) * Bt[N,K](bf16)^T + bias
// 128x128 tile, BK=32, 8 waves (4 in M x 2 in N), each wave 32x64.
// Double-buffered LDS, filled with async global->LDS DMA (ASYNCcnt).
template <bool OUT_BF16>
__global__ __launch_bounds__(256) void gemm_kernel(
    const void* __restrict__ A_, const void* __restrict__ Bt_,
    const float* __restrict__ bias, void* __restrict__ C_,
    int M, int N, int K) {
  const bf16_t* A  = (const bf16_t*)A_;
  const bf16_t* Bt = (const bf16_t*)Bt_;
  __shared__ __attribute__((aligned(16))) bf16_t As[2][128][32];
  __shared__ __attribute__((aligned(16))) bf16_t Bs[2][128][32];

  const int tid  = threadIdx.x;
  const int wave = tid >> 5;
  const int lane = tid & 31;
  const int half = lane >> 4;
  const int ln16 = lane & 15;
  const int wm   = wave & 3;   // M group (32 rows each)
  const int wn   = wave >> 2;  // N group (64 cols each)
  const int tileM = blockIdx.x * 128;
  const int tileN = blockIdx.y * 128;

  v8f acc[2][4];
#pragma unroll
  for (int mi = 0; mi < 2; ++mi)
#pragma unroll
    for (int ni = 0; ni < 4; ++ni) acc[mi][ni] = {};

  auto issue_tile = [&](int buf, int k0) {
#pragma unroll
    for (int it = 0; it < 2; ++it) {
      int idx = tid + it * 256;          // 0..511 16B chunks per matrix
      int row = idx >> 2;
      int chk = idx & 3;
      async_g2l_b128(lds_off(&As[buf][row][chk * 8]),
                     A + (size_t)(tileM + row) * K + k0 + chk * 8);
      async_g2l_b128(lds_off(&Bs[buf][row][chk * 8]),
                     Bt + (size_t)(tileN + row) * K + k0 + chk * 8);
    }
  };

  const int nk = K >> 5;
  issue_tile(0, 0);
  for (int kk = 0; kk < nk; ++kk) {
    const int cur = kk & 1;
    wait_async0();
    __syncthreads();                     // tile `cur` visible to all waves
    if (kk + 1 < nk) issue_tile(1 - cur, (kk + 1) * 32);
    if (kk + 2 < nk) {                   // prefetch 2 tiles ahead into L2
      __builtin_prefetch(A + (size_t)(tileM + (tid >> 1)) * K + (kk + 2) * 32, 0, 1);
      __builtin_prefetch(Bt + (size_t)(tileN + (tid >> 1)) * K + (kk + 2) * 32, 0, 1);
    }

    v16bf af[2], bf_[4];
#pragma unroll
    for (int mi = 0; mi < 2; ++mi)
      af[mi] = load_frag(&As[cur][wm * 32 + mi * 16 + ln16][0], half);
#pragma unroll
    for (int ni = 0; ni < 4; ++ni)
      bf_[ni] = load_frag(&Bs[cur][wn * 64 + ni * 16 + ln16][0], half);
#pragma unroll
    for (int mi = 0; mi < 2; ++mi)
#pragma unroll
      for (int ni = 0; ni < 4; ++ni)
        acc[mi][ni] = wmma_bf16(af[mi], bf_[ni], acc[mi][ni]);
    __syncthreads();                     // all reads of `cur` done
  }

#pragma unroll
  for (int mi = 0; mi < 2; ++mi)
#pragma unroll
    for (int ni = 0; ni < 4; ++ni) {
      int col = tileN + wn * 64 + ni * 16 + ln16;
      float bb = bias[col];
#pragma unroll
      for (int i = 0; i < 8; ++i) {
        int row = tileM + wm * 32 + mi * 16 + i + 8 * half;
        float v = acc[mi][ni][i] + bb;
        if (OUT_BF16) ((bf16_t*)C_)[(size_t)row * N + col] = (bf16_t)v;
        else          ((float*)C_)[(size_t)row * N + col]  = v;
      }
    }
}

// ================================================================ Flash attention
// Q,K: (B,H,S,HD) bf16 (Q pre-scaled by 1/sqrt(HD)); Vt: (B,H,HD,S) bf16.
// Grid (S/128, B*H), 8 waves, wave = 16 query rows x full HD=128 output.
// K/V tiles double-buffered in LDS via async DMA.
__global__ __launch_bounds__(256) void flash_kernel(
    const void* __restrict__ Q_, const void* __restrict__ K_,
    const void* __restrict__ Vt_, void* __restrict__ O_) {
  const bf16_t* Q  = (const bf16_t*)Q_;
  const bf16_t* Km = (const bf16_t*)K_;
  const bf16_t* Vt = (const bf16_t*)Vt_;
  bf16_t* O = (bf16_t*)O_;

  __shared__ __attribute__((aligned(16))) bf16_t Ks[2][32][128];   // keys x hd
  __shared__ __attribute__((aligned(16))) bf16_t Vs[2][128][32];   // hd x keys
  __shared__ __attribute__((aligned(16))) bf16_t Ps[8][16][32];    // wave-private P

  const int tid  = threadIdx.x;
  const int wave = tid >> 5;
  const int lane = tid & 31;
  const int half = lane >> 4;
  const int ln16 = lane & 15;
  const int bh = blockIdx.y;           // b*H + h
  const int b = bh >> 4, h = bh & 15;
  const int q0 = blockIdx.x * 128 + wave * 16;

  const bf16_t* qbase = Q + ((size_t)bh * Sc + q0) * HDc;
  const bf16_t* kbase = Km + (size_t)bh * Sc * HDc;
  const bf16_t* vbase = Vt + (size_t)bh * HDc * Sc;

  v16bf qf[4];
#pragma unroll
  for (int ks = 0; ks < 4; ++ks)
    qf[ks] = load_frag(qbase + (size_t)ln16 * HDc + ks * 32, half);

  v8f o_[8];
#pragma unroll
  for (int ni = 0; ni < 8; ++ni) o_[ni] = {};
  float m_[8], l_[8];
#pragma unroll
  for (int i = 0; i < 8; ++i) { m_[i] = -1e30f; l_[i] = 0.f; }

  auto issue_kv = [&](int buf, int k0) {
#pragma unroll
    for (int it = 0; it < 2; ++it) {
      int idx = tid + it * 256;
      int kr = idx >> 4, kc = idx & 15;  // Ks: 32 rows x 256B
      async_g2l_b128(lds_off(&Ks[buf][kr][kc * 8]),
                     kbase + (size_t)(k0 + kr) * HDc + kc * 8);
      int vr = idx >> 2, vc = idx & 3;   // Vs: 128 rows x 64B
      async_g2l_b128(lds_off(&Vs[buf][vr][vc * 8]),
                     vbase + (size_t)vr * Sc + k0 + vc * 8);
    }
  };

  const int nk = Sc / 32;
  issue_kv(0, 0);
  for (int kk = 0; kk < nk; ++kk) {
    const int cur = kk & 1;
    wait_async0();
    __syncthreads();
    if (kk + 1 < nk) issue_kv(1 - cur, (kk + 1) * 32);

    v8f s0 = {}, s1 = {};
#pragma unroll
    for (int ks = 0; ks < 4; ++ks) {
      v16bf b0 = load_frag(&Ks[cur][ln16][ks * 32], half);        // keys 0..15
      v16bf b1 = load_frag(&Ks[cur][16 + ln16][ks * 32], half);   // keys 16..31
      s0 = wmma_bf16(qf[ks], b0, s0);
      s1 = wmma_bf16(qf[ks], b1, s1);
    }

    // online softmax per C-vgpr row (row = i + 8*half, cols striped over 16 lanes)
#pragma unroll
    for (int i = 0; i < 8; ++i) {
      float mt = fmaxf(s0[i], s1[i]);
      mt = fmaxf(mt, __shfl_xor(mt, 1, 32));
      mt = fmaxf(mt, __shfl_xor(mt, 2, 32));
      mt = fmaxf(mt, __shfl_xor(mt, 4, 32));
      mt = fmaxf(mt, __shfl_xor(mt, 8, 32));
      float mn = fmaxf(m_[i], mt);
      float sc = __expf(m_[i] - mn);
      float p0 = __expf(s0[i] - mn);
      float p1 = __expf(s1[i] - mn);
      float ps = p0 + p1;
      ps += __shfl_xor(ps, 1, 32);
      ps += __shfl_xor(ps, 2, 32);
      ps += __shfl_xor(ps, 4, 32);
      ps += __shfl_xor(ps, 8, 32);
      l_[i] = l_[i] * sc + ps;
      m_[i] = mn;
#pragma unroll
      for (int ni = 0; ni < 8; ++ni) o_[ni][i] *= sc;
      Ps[wave][i + 8 * half][ln16]      = (bf16_t)p0;
      Ps[wave][i + 8 * half][16 + ln16] = (bf16_t)p1;
    }

    v16bf pf = load_frag(&Ps[wave][ln16][0], half);
#pragma unroll
    for (int ni = 0; ni < 8; ++ni) {
      v16bf vf = load_frag(&Vs[cur][ni * 16 + ln16][0], half);
      o_[ni] = wmma_bf16(pf, vf, o_[ni]);
    }
    __syncthreads();
  }

  // write O to (B,S,D) layout (col = h*HD + hd)
#pragma unroll
  for (int ni = 0; ni < 8; ++ni)
#pragma unroll
    for (int i = 0; i < 8; ++i) {
      int rl = i + 8 * half;
      float v = o_[ni][i] / l_[i];
      O[((size_t)(b * Sc + q0 + rl)) * Dc + h * HDc + ni * 16 + ln16] = (bf16_t)v;
    }
}

// ================================================================ SwiGLU elementwise: g = silu(h1)*h3
__global__ __launch_bounds__(256) void swiglu_kernel(
    const void* __restrict__ h1_, const void* __restrict__ h3_,
    void* __restrict__ g_, long long n) {
  long long i = (long long)blockIdx.x * 256 + threadIdx.x;
  if (i < n) {
    float a = (float)((const bf16_t*)h1_)[i];
    float c = (float)((const bf16_t*)h3_)[i];
    float g = (a / (1.f + __expf(-a))) * c;
    ((bf16_t*)g_)[i] = (bf16_t)g;
  }
}

// ================================================================ host orchestration
extern "C" void kernel_launch(void* const* d_in, const int* in_sizes, int n_in,
                              void* d_out, int out_size, void* d_ws, size_t ws_size,
                              hipStream_t stream) {
  (void)in_sizes; (void)n_in; (void)out_size; (void)ws_size;
  const long long BS = (long long)Bc * Sc;             // 4096
  const size_t actB = (size_t)BS * Dc * 2;             // bf16 activation (B,S,D)
  const size_t DD2  = (size_t)Dc * Dc * 2;
  const size_t DH2  = (size_t)Dc * HIDc * 2;

  char* ws = (char*)d_ws;
  size_t p = 0;
  auto take = [&](size_t bytes) {
    p = (p + 255) & ~(size_t)255;
    size_t o = p; p += bytes; return o;
  };
  void* nx   = ws + take(actB);
  void* ny   = ws + take(actB);
  void* wqT  = ws + take(DD2);
  void* wkT  = ws + take(DD2);
  void* wvT  = ws + take(DD2);
  void* woT  = ws + take(DD2);
  void* w1T  = ws + take(DH2);
  void* w3T  = ws + take(DH2);
  void* w2T  = ws + take(DH2);
  void* qlin = ws + take(actB);
  void* klin = ws + take(actB);
  void* vlin = ws + take(actB);
  void* Qr   = ws + take(actB);
  void* Kr   = ws + take(actB);
  void* Vtb  = ws + take(actB);
  void* attn = ws + take(actB);
  void* hb   = ws + take(actB);
  void* h1   = ws + take((size_t)BS * HIDc * 2);
  void* h3   = ws + take((size_t)BS * HIDc * 2);
  void* ff   = ws + take((size_t)BS * Dc * 4);

  const float* x   = (const float*)d_in[0];
  const float* y   = (const float*)d_in[1];
  const float* cs  = (const float*)d_in[3];
  const float* sn  = (const float*)d_in[4];
  const float* anw = (const float*)d_in[5];
  float* out = (float*)d_out;

  dim3 blk(256);
  // initial rmsnorm -> bf16
  rmsnorm_kernel<true, false><<<dim3(BS), blk, 0, stream>>>(x, nullptr, anw, nx);
  rmsnorm_kernel<true, false><<<dim3(BS), blk, 0, stream>>>(y, nullptr, anw, ny);

  const float qscale = 0.08838834764831845f;  // 1/sqrt(128)

  for (int side = 0; side < 2; ++side) {
    const int o = 6 + 15 * side;
    const float* Wq = (const float*)d_in[o + 0];
    const float* bq = (const float*)d_in[o + 1];
    const float* Wk = (const float*)d_in[o + 2];
    const float* bk = (const float*)d_in[o + 3];
    const float* Wv = (const float*)d_in[o + 4];
    const float* bv = (const float*)d_in[o + 5];
    const float* Wo = (const float*)d_in[o + 6];
    const float* bo = (const float*)d_in[o + 7];
    const float* W1 = (const float*)d_in[o + 8];
    const float* b1 = (const float*)d_in[o + 9];
    const float* W3 = (const float*)d_in[o + 10];
    const float* b3 = (const float*)d_in[o + 11];
    const float* W2 = (const float*)d_in[o + 12];
    const float* b2 = (const float*)d_in[o + 13];
    const float* fnw = (const float*)d_in[o + 14];

    const void* context = (side == 0) ? nx : ny;  // k,v source
    const void* hidden  = (side == 0) ? ny : nx;  // q source
    const float* resid  = (side == 0) ? x : y;

    // weight transpose+cast
    tcast_kernel<<<dim3(Dc / 32, Dc / 32), blk, 0, stream>>>(Wq, wqT, Dc, Dc);
    tcast_kernel<<<dim3(Dc / 32, Dc / 32), blk, 0, stream>>>(Wk, wkT, Dc, Dc);
    tcast_kernel<<<dim3(Dc / 32, Dc / 32), blk, 0, stream>>>(Wv, wvT, Dc, Dc);
    tcast_kernel<<<dim3(Dc / 32, Dc / 32), blk, 0, stream>>>(Wo, woT, Dc, Dc);
    tcast_kernel<<<dim3(Dc / 32, HIDc / 32), blk, 0, stream>>>(W1, w1T, Dc, HIDc);
    tcast_kernel<<<dim3(Dc / 32, HIDc / 32), blk, 0, stream>>>(W3, w3T, Dc, HIDc);
    tcast_kernel<<<dim3(HIDc / 32, Dc / 32), blk, 0, stream>>>(W2, w2T, HIDc, Dc);

    // projections (bf16 out)
    gemm_kernel<true><<<dim3(BS / 128, Dc / 128), blk, 0, stream>>>(hidden,  wqT, bq, qlin, BS, Dc, Dc);
    gemm_kernel<true><<<dim3(BS / 128, Dc / 128), blk, 0, stream>>>(context, wkT, bk, klin, BS, Dc, Dc);
    gemm_kernel<true><<<dim3(BS / 128, Dc / 128), blk, 0, stream>>>(context, wvT, bv, vlin, BS, Dc, Dc);

    // rope reshape + V transpose
    rope_kernel<<<dim3((Bc * Sc * Hc * (HDc / 2)) / 256), blk, 0, stream>>>(qlin, cs, sn, Qr, qscale);
    rope_kernel<<<dim3((Bc * Sc * Hc * (HDc / 2)) / 256), blk, 0, stream>>>(klin, cs, sn, Kr, 1.0f);
    vtile_kernel<<<dim3(Sc / 32, HDc / 32, Bc * Hc), blk, 0, stream>>>(vlin, Vtb);

    // flash attention
    flash_kernel<<<dim3(Sc / 128, Bc * Hc), blk, 0, stream>>>(Qr, Kr, Vtb, attn);

    // output projection (bf16 out -> ffn input)
    gemm_kernel<true><<<dim3(BS / 128, Dc / 128), blk, 0, stream>>>(attn, woT, bo, hb, BS, Dc, Dc);

    // FFN
    gemm_kernel<true><<<dim3(BS / 128, HIDc / 128), blk, 0, stream>>>(hb, w1T, b1, h1, BS, HIDc, Dc);
    gemm_kernel<true><<<dim3(BS / 128, HIDc / 128), blk, 0, stream>>>(hb, w3T, b3, h3, BS, HIDc, Dc);
    swiglu_kernel<<<dim3((unsigned)((BS * HIDc) / 256)), blk, 0, stream>>>(h1, h3, h1, BS * (long long)HIDc);
    gemm_kernel<false><<<dim3(BS / 128, Dc / 128), blk, 0, stream>>>(h1, w2T, b2, ff, BS, Dc, HIDc);

    // final residual + rmsnorm -> f32 output
    rmsnorm_kernel<false, true><<<dim3(BS), blk, 0, stream>>>(
        (const float*)ff, resid, fnw, out + (size_t)side * BS * Dc);
  }
}
